// VPEncoderLayer_48146583388539
// MI455X (gfx1250) — matrix operand order repaired and checked by reading, not compiled, not run on hardware
//
#include <hip/hip_runtime.h>
#include <hip/hip_bf16.h>

// ---------------------------------------------------------------------------
// VPEncoderLayer (bilinear cross-attention) for MI455X / gfx1250, wave32.
// All matmuls via v_wmma_f32_16x16x32_f16 (f16 in, fp32 accumulate).
// Flash-style fused attention: scores never touch HBM.
// Async global->LDS staging (ASYNCcnt) where the copy is transpose-free.
//
// Workspace layout (needs 32 MiB):
//   [ 0M, 4M)  xq  f16  [8192,256]   x + x_embedding
//   [ 4M, 8M)  sk  f16  [8192,256]   source1 + source_embedding
//   [ 8M,12M)  q   f16  [8192,256]   (pre-scaled by 1/sqrt(D)*log2(e))
//   [12M,16M)  k   f16  [8192,256]
//   [16M,20M)  v2  f16  [8192,256]
//   [20M,28M)  v1  f32  [8192,256]
//   [28M,32M)  msg f16  [8192,256]
// ---------------------------------------------------------------------------

typedef _Float16 f16_t;
typedef __attribute__((ext_vector_type(16))) _Float16 v16h;
typedef __attribute__((ext_vector_type(8)))  float    v8f;

#define GLOBAL_AS __attribute__((address_space(1)))
#define LDS_AS    __attribute__((address_space(3)))

// Matches the builtin's parameter type per hipcc diagnostic:
//   __attribute__((__vector_size__(4 * sizeof(int)))) int __device__ *
typedef int avec4 __attribute__((vector_size(4 * sizeof(int))));

#if defined(__has_builtin)
#if __has_builtin(__builtin_amdgcn_global_load_async_to_lds_b128) && \
    __has_builtin(__builtin_amdgcn_s_wait_asynccnt)
#define HAVE_ASYNC_LDS 1
#endif
#endif
#ifndef HAVE_ASYNC_LDS
#define HAVE_ASYNC_LDS 0
#endif

union FragH {            // 8 VGPRs = 16 packed f16 (WMMA A/B operand)
    unsigned int u[8];
    v16h v;
};

__device__ __forceinline__ unsigned short f2h_bits(float f) {
    _Float16 h = (_Float16)f;
    return __builtin_bit_cast(unsigned short, h);
}

// async (or fallback) 16-byte global -> LDS copy
__device__ __forceinline__ void copy16_g2lds(const uint4* g, uint4* lds) {
#if HAVE_ASYNC_LDS
    __builtin_amdgcn_global_load_async_to_lds_b128(
        (GLOBAL_AS avec4*)g, (LDS_AS avec4*)lds, 0, 0);
#else
    *lds = *g;
#endif
}

__device__ __forceinline__ void wait_async_lds() {
#if HAVE_ASYNC_LDS
    __builtin_amdgcn_s_wait_asynccnt(0);
#endif
}

// ---------------------------------------------------------------------------
// Kernel 1: out_f16[i] = f16(a[i] + b[i]), vectorized by 4.
// ---------------------------------------------------------------------------
__global__ __launch_bounds__(256)
void add_convert_kernel(const float* __restrict__ a, const float* __restrict__ b,
                        f16_t* __restrict__ out, int n4) {
    int i = blockIdx.x * 256 + threadIdx.x;
    if (i >= n4) return;
    float4 x = ((const float4*)a)[i];
    float4 y = ((const float4*)b)[i];
    unsigned int lo = (unsigned int)f2h_bits(x.x + y.x) |
                      ((unsigned int)f2h_bits(x.y + y.y) << 16);
    unsigned int hi = (unsigned int)f2h_bits(x.z + y.z) |
                      ((unsigned int)f2h_bits(x.w + y.w) << 16);
    uint2 pk; pk.x = lo; pk.y = hi;
    ((uint2*)out)[i] = pk;
}

// ---------------------------------------------------------------------------
// Kernel 2: Y = X @ W^T  (X: [M,256] f16 row-major, W: [256,256] fp32 row-major)
// Block: 128 threads (4 waves). Block tile 64(M) x 64(N), K stepped by 32.
// OUT_F32: write fp32 (+bias if non-null); else write f16 scaled by oscale.
// ---------------------------------------------------------------------------
template <bool OUT_F32>
__global__ __launch_bounds__(128)
void gemm_xwT_kernel(const f16_t* __restrict__ X, const float* __restrict__ W,
                     const float* __restrict__ bias,
                     f16_t* __restrict__ Yh, float* __restrict__ Yf,
                     float oscale) {
    __shared__ unsigned int xs[64 * 20];   // X tile [64 m][32 k] f16, 20 dw/row (16B rows)
    __shared__ unsigned int wsm[32 * 33];  // W^T tile [32 k][64 n] f16, 33 dw/row

    const int t    = threadIdx.x;
    const int lane = t & 31;
    const int wv   = t >> 5;
    const int half = lane >> 4;
    const int l15  = lane & 15;
    const int m0   = blockIdx.x * 64;
    const int n0   = blockIdx.y * 64;

    v8f acc[4] = {};

    const unsigned int* Xg = (const unsigned int*)X;  // f16 pairs

    for (int k0 = 0; k0 < 256; k0 += 32) {
        // ---- stage X tile: 256 x 16B segments, 2 per thread (async to LDS) ----
        #pragma unroll
        for (int u = 0; u < 2; ++u) {
            int idx = t * 2 + u;            // 0..255
            int row = idx >> 2, q4 = idx & 3;
            copy16_g2lds((const uint4*)&Xg[(size_t)(m0 + row) * 128 + (k0 >> 1) + q4 * 4],
                         (uint4*)&xs[row * 20 + q4 * 4]);
        }
        // ---- stage W tile transposed + fp32->f16: 2048 elems, 16 per thread ----
        unsigned short* wss = (unsigned short*)wsm;
        #pragma unroll
        for (int u = 0; u < 16; ++u) {
            int idx = t * 16 + u;
            int j = idx >> 5, k = idx & 31;
            float w = W[(size_t)(n0 + j) * 256 + k0 + k];
            wss[k * 66 + j] = f2h_bits(w);
        }
        if (k0 + 32 < 256)   // prefetch next W tile line
            __builtin_prefetch(&W[(size_t)(n0 + (t >> 1)) * 256 + k0 + 32 + (t & 1) * 16], 0, 1);
        wait_async_lds();
        __syncthreads();

        // ---- A fragment: rows m0+16*wv+l15, K pattern per ISA 16-bit A layout ----
        FragH a;
        const int arow = wv * 16 + l15;
        #pragma unroll
        for (int i = 0; i < 8; ++i) {
            int kp = ((i < 4) ? (2 * i) : (16 + 2 * (i - 4))) + half * 8;
            a.u[i] = xs[arow * 20 + (kp >> 1)];
        }
        // ---- 4 N-tiles: B column n = l15, K pattern same as A ----
        #pragma unroll
        for (int nt = 0; nt < 4; ++nt) {
            FragH b;
            const unsigned short* wr = (const unsigned short*)wsm + nt * 16 + l15;
            #pragma unroll
            for (int i = 0; i < 8; ++i) {
                int kp = ((i < 4) ? (2 * i) : (16 + 2 * (i - 4))) + half * 8;
                b.u[i] = (unsigned int)wr[kp * 66] | ((unsigned int)wr[(kp + 1) * 66] << 16);
            }
            acc[nt] = __builtin_amdgcn_wmma_f32_16x16x32_f16(
                false, a.v, false, b.v, (short)0, acc[nt], false, false);
        }
        __syncthreads();
    }

    // ---- epilogue: C layout -> memory ----
    #pragma unroll
    for (int nt = 0; nt < 4; ++nt) {
        #pragma unroll
        for (int r = 0; r < 8; ++r) {
            int grow = m0 + wv * 16 + r + half * 8;
            int gcol = n0 + nt * 16 + l15;
            float val = acc[nt][r];
            if (OUT_F32) {
                Yf[(size_t)grow * 256 + gcol] = val + (bias ? bias[gcol] : 0.0f);
            } else {
                Yh[(size_t)grow * 256 + gcol] = (f16_t)(val * oscale);
            }
        }
    }
}

// ---------------------------------------------------------------------------
// Kernel 3: fused bilinear attention (flash-style, softmax over S in registers)
//   q arrives pre-scaled by (1/sqrt(D))*log2(e)  ->  softmax via exp2.
// Grid: (L/64, N*H). Block: 128 threads = 4 waves; wave owns 16 L-rows.
// ---------------------------------------------------------------------------
__global__ __launch_bounds__(128)
void attn_kernel(const f16_t* __restrict__ Q, const f16_t* __restrict__ K,
                 const f16_t* __restrict__ V2, const float* __restrict__ V1,
                 f16_t* __restrict__ MSG) {
    constexpr int L = 1024, S = 1024, C = 256, D = 32;

    __shared__ unsigned int kts[32 * 33];      // K^T chunk [32 d][64 s], padded
    __shared__ unsigned int v2s[64 * 20];      // V2 chunk  [64 s][32 d], 16B rows
    __shared__ unsigned int pw[4][16 * 33];    // per-wave P scratch [16 l][64 s]

    const int t    = threadIdx.x;
    const int lane = t & 31;
    const int wv   = t >> 5;
    const int half = lane >> 4;
    const int l15  = lane & 15;
    const int n    = blockIdx.y >> 3;
    const int h    = blockIdx.y & 7;
    const int r0   = blockIdx.x * 64 + wv * 16;   // first L row of this wave

    // ---- persistent Q fragment (A operand, 16 rows x 32 d) ----
    FragH qa;
    {
        const f16_t* qp = Q + ((size_t)(n * L + r0 + l15) * C + h * D);
        uint4 lo = *(const uint4*)(qp + half * 8);        // K 0..7  / 8..15
        uint4 hi = *(const uint4*)(qp + 16 + half * 8);   // K 16..23 / 24..31
        qa.u[0] = lo.x; qa.u[1] = lo.y; qa.u[2] = lo.z; qa.u[3] = lo.w;
        qa.u[4] = hi.x; qa.u[5] = hi.y; qa.u[6] = hi.z; qa.u[7] = hi.w;
    }

    float mrow[8], lrow[8];
    #pragma unroll
    for (int r = 0; r < 8; ++r) { mrow[r] = -1.0e30f; lrow[r] = 0.0f; }
    v8f o0 = {}, o1 = {};

    for (int c = 0; c < S / 64; ++c) {
        const int s0 = c * 64;
        // ---- stage V2 chunk: 256 x 16B segments, 2 per thread (async) ----
        #pragma unroll
        for (int u = 0; u < 2; ++u) {
            int idx = t * 2 + u;            // 0..255
            int s = idx >> 2, q4 = idx & 3;
            size_t g = (size_t)(n * S + s0 + s) * C + h * D + q4 * 8;
            copy16_g2lds((const uint4*)(V2 + g), (uint4*)&v2s[s * 20 + q4 * 4]);
        }
        // ---- stage K^T chunk (transpose through shorts): 1024 dwords ----
        unsigned short* ktss = (unsigned short*)kts;
        #pragma unroll
        for (int u = 0; u < 8; ++u) {
            int idx = t * 8 + u;
            int s = idx >> 4, dw = idx & 15;
            size_t g = (size_t)(n * S + s0 + s) * C + h * D;
            unsigned int kd = ((const unsigned int*)(K + g))[dw];
            ktss[(2 * dw) * 66 + s]     = (unsigned short)(kd & 0xffffu);
            ktss[(2 * dw + 1) * 66 + s] = (unsigned short)(kd >> 16);
        }
        wait_async_lds();
        __syncthreads();

        // ---- scores: 4 tiles of 16x16, full D in one WMMA each ----
        v8f sc[4];
        #pragma unroll
        for (int st = 0; st < 4; ++st) {
            FragH b;
            const unsigned short* kr = (const unsigned short*)kts + st * 16 + l15;
            #pragma unroll
            for (int i = 0; i < 8; ++i) {
                int kp = ((i < 4) ? (2 * i) : (16 + 2 * (i - 4))) + half * 8;
                b.u[i] = (unsigned int)kr[kp * 66] | ((unsigned int)kr[(kp + 1) * 66] << 16);
            }
            v8f z = {};
            sc[st] = __builtin_amdgcn_wmma_f32_16x16x32_f16(
                false, qa.v, false, b.v, (short)0, z, false, false);
        }

        // ---- online softmax update (base-2; scale folded into q) ----
        float mnew[8], pscale[8];
        #pragma unroll
        for (int r = 0; r < 8; ++r) {
            float mx = fmaxf(fmaxf(sc[0][r], sc[1][r]), fmaxf(sc[2][r], sc[3][r]));
            mx = fmaxf(mx, __shfl_xor(mx, 1, 16));
            mx = fmaxf(mx, __shfl_xor(mx, 2, 16));
            mx = fmaxf(mx, __shfl_xor(mx, 4, 16));
            mx = fmaxf(mx, __shfl_xor(mx, 8, 16));
            mnew[r]   = fmaxf(mrow[r], mx);
            pscale[r] = exp2f(mrow[r] - mnew[r]);
            mrow[r]   = mnew[r];
        }
        #pragma unroll
        for (int r = 0; r < 8; ++r) {
            float s = 0.0f;
            #pragma unroll
            for (int st = 0; st < 4; ++st) {
                float p = exp2f(sc[st][r] - mnew[r]);
                sc[st][r] = p;
                s += p;
            }
            s += __shfl_xor(s, 1, 16);
            s += __shfl_xor(s, 2, 16);
            s += __shfl_xor(s, 4, 16);
            s += __shfl_xor(s, 8, 16);
            lrow[r] = lrow[r] * pscale[r] + s;
            o0[r] *= pscale[r];
            o1[r] *= pscale[r];
        }

        // ---- remap P: C layout -> A layout via per-wave LDS scratch ----
        unsigned short* pws = (unsigned short*)pw[wv];
        #pragma unroll
        for (int st = 0; st < 4; ++st)
            #pragma unroll
            for (int r = 0; r < 8; ++r)
                pws[(r + half * 8) * 66 + st * 16 + l15] = f2h_bits(sc[st][r]);

        // ---- O += P @ V2 : 2 K-halves x 2 D-tiles ----
        #pragma unroll
        for (int kk = 0; kk < 2; ++kk) {
            FragH pa;
            #pragma unroll
            for (int i = 0; i < 8; ++i) {
                int kp = kk * 32 + ((i < 4) ? (2 * i) : (16 + 2 * (i - 4))) + half * 8;
                pa.u[i] = pw[wv][l15 * 33 + (kp >> 1)];
            }
            #pragma unroll
            for (int dt = 0; dt < 2; ++dt) {
                FragH b;
                const unsigned short* vr = (const unsigned short*)v2s + dt * 16 + l15;
                #pragma unroll
                for (int i = 0; i < 8; ++i) {
                    int kp = kk * 32 + ((i < 4) ? (2 * i) : (16 + 2 * (i - 4))) + half * 8;
                    b.u[i] = (unsigned int)vr[kp * 40] | ((unsigned int)vr[(kp + 1) * 40] << 16);
                }
                if (dt == 0)
                    o0 = __builtin_amdgcn_wmma_f32_16x16x32_f16(
                        false, pa.v, false, b.v, (short)0, o0, false, false);
                else
                    o1 = __builtin_amdgcn_wmma_f32_16x16x32_f16(
                        false, pa.v, false, b.v, (short)0, o1, false, false);
            }
        }
        __syncthreads();
    }

    // ---- epilogue: msg = v1 * (O / l) ----
    #pragma unroll
    for (int r = 0; r < 8; ++r) {
        int row = r0 + r + half * 8;
        float inv = 1.0f / lrow[r];
        size_t base = (size_t)(n * L + row) * C + h * D;
        float g0 = V1[base + l15];
        float g1 = V1[base + 16 + l15];
        MSG[base + l15]      = (f16_t)(o0[r] * inv * g0);
        MSG[base + 16 + l15] = (f16_t)(o1[r] * inv * g1);
    }
}

// ---------------------------------------------------------------------------
extern "C" void kernel_launch(void* const* d_in, const int* in_sizes, int n_in,
                              void* d_out, int out_size, void* d_ws, size_t ws_size,
                              hipStream_t stream) {
    const float* x  = (const float*)d_in[0];
    const float* s1 = (const float*)d_in[1];
    const float* xe = (const float*)d_in[2];
    const float* se = (const float*)d_in[3];
    const float* Wq = (const float*)d_in[4];
    const float* Wk = (const float*)d_in[5];
    const float* Wv = (const float*)d_in[6];
    const float* Wp = (const float*)d_in[7];
    const float* bp = (const float*)d_in[8];
    float* out = (float*)d_out;

    char* w = (char*)d_ws;                       // needs 32 MiB of scratch
    const size_t MB = (size_t)1 << 20;
    f16_t* xq = (f16_t*)(w + 0 * MB);
    f16_t* sk = (f16_t*)(w + 4 * MB);
    f16_t* qb = (f16_t*)(w + 8 * MB);
    f16_t* kb = (f16_t*)(w + 12 * MB);
    f16_t* v2 = (f16_t*)(w + 16 * MB);
    float* v1 = (float*)(w + 20 * MB);
    f16_t* mg = (f16_t*)(w + 28 * MB);

    const int M = 8 * 1024;                      // N*L == N*S == 8192 rows

    // 1) embedding add + f16 convert
    int n4 = M * 256 / 4;
    add_convert_kernel<<<n4 / 256, 256, 0, stream>>>(x, xe, xq, n4);
    add_convert_kernel<<<n4 / 256, 256, 0, stream>>>(s1, se, sk, n4);

    // 2) projections (q pre-scaled by 1/sqrt(D)*log2(e); k,v2 f16; v1 fp32)
    const float qscale = 0.17677669529663687f * 1.4426950408889634f;
    dim3 gg(M / 64, 256 / 64);
    gemm_xwT_kernel<false><<<gg, 128, 0, stream>>>(xq, Wq, nullptr, qb, nullptr, qscale);
    gemm_xwT_kernel<false><<<gg, 128, 0, stream>>>(sk, Wk, nullptr, kb, nullptr, 1.0f);
    gemm_xwT_kernel<true ><<<gg, 128, 0, stream>>>(xq, Wv, nullptr, nullptr, v1, 1.0f);
    gemm_xwT_kernel<false><<<gg, 128, 0, stream>>>(sk, Wv, nullptr, v2, nullptr, 1.0f);

    // 3) fused flash attention + v1 gate
    attn_kernel<<<dim3(1024 / 64, 8 * 8), 128, 0, stream>>>(qb, kb, v2, v1, mg);

    // 4) output projection (+bias) -> fp32 d_out
    gemm_xwT_kernel<true><<<gg, 128, 0, stream>>>(mg, Wp, bp, nullptr, out, 1.0f);
}
